// LoopRelationalGraphConvolution_77386720740027
// MI455X (gfx1250) — matrix operand order repaired and checked by reading, not compiled, not run on hardware
//
#include <hip/hip_runtime.h>
#include <stdint.h>

// ---------------- types ----------------
typedef __attribute__((ext_vector_type(16))) _Float16 v16h;
typedef __attribute__((ext_vector_type(8)))  _Float16 v8h;
typedef __attribute__((ext_vector_type(8)))  float    v8f;
typedef __attribute__((ext_vector_type(4)))  float    v4f;

// ---------------- problem constants ----------------
#define D_IN        256
#define D_OUT       256
#define S_EDGES     32
#define R1          33          // NUM_REL + 1
#define NODES_PER_BLOCK 64
#define CR          4           // relations per LDS chunk
#define KTILES      8           // 256 / 32 (K per WMMA)
#define NTILES      16          // 256 / 16
#define LDS_BYTES   (CR * NODES_PER_BLOCK * D_IN * 2 + NODES_PER_BLOCK * S_EDGES * 8)
// bytes of one relation's packed B (8 kTiles x 16 nTiles x 32 lanes x 32B)
#define B_REL_BYTES (KTILES * NTILES * 32 * 32)

// =====================================================================
// Kernel 1: prepack f32 weights [33][256][256] into f16 WMMA-B layout.
// B = W_r^T  (B[K=d][N=o] = W[r][o][d]).
// Per B tile (one r, one 32-K slice, one 16-N slice): 32 lanes x 16 f16.
//   lane L: N = nt*16 + (L & 15),  K = kt*32 + (L>>4)*16 + j  (j = 0..15)
// Output index: (((r*8 + kt)*16 + nt)*32 + lane)*16 halves -> 32B per lane,
// so the main kernel's B load is one 32B per-lane load at an immediate offset.
// =====================================================================
__global__ __launch_bounds__(256) void prepack_weights_kernel(
    const float* __restrict__ W, _Float16* __restrict__ wB)
{
    int t = blockIdx.x * blockDim.x + threadIdx.x;   // (r, kt, nt, lane)
    int lane = t & 31;
    int nt   = (t >> 5) & 15;
    int kt   = (t >> 9) & 7;
    int r    = t >> 12;
    if (r >= R1) return;
    int n     = nt * 16 + (lane & 15);
    int kbase = kt * 32 + (lane >> 4) * 16;
    const float* src = W + ((size_t)r * D_OUT + n) * D_IN + kbase;
    _Float16*    dst = wB + (size_t)t * 16;
#pragma unroll
    for (int j = 0; j < 16; ++j) dst[j] = (_Float16)src[j];
}

// =====================================================================
// Kernel 2: fused gather -> per-relation mean -> batched GEMM -> relu.
// Block: 64 nodes, 256 threads (8 wave32 waves).
// =====================================================================
__global__ __launch_bounds__(256) void rgcn_kernel(
    const float*    __restrict__ emb,        // [V, 256] f32
    const _Float16* __restrict__ wB,         // prepacked B
    const int*      __restrict__ neighbors,  // [8192, 32]
    const int*      __restrict__ relations,  // [8192, 32]
    float*          __restrict__ out)        // [8192, 256]
{
    extern __shared__ char smem[];
    _Float16* sAgg = (_Float16*)smem;                              // [CR][64][256] f16
    int* sNbr = (int*)(smem + (size_t)CR * NODES_PER_BLOCK * D_IN * 2);  // [2048]
    int* sRel = sNbr + NODES_PER_BLOCK * S_EDGES;                        // [2048]

    const int tid  = threadIdx.x;
    const int lane = tid & 31;
    const int wave = tid >> 5;
    const int blockNode0 = blockIdx.x * NODES_PER_BLOCK;

    // Preload this block's edge lists into LDS (read once, scanned per chunk).
    for (int i = tid; i < NODES_PER_BLOCK * S_EDGES; i += 256) {
        sNbr[i] = neighbors[(size_t)blockNode0 * S_EDGES + i];
        sRel[i] = relations[(size_t)blockNode0 * S_EDGES + i];
    }

    // Wave -> (m-group, n-group): 2 x 4. Wave covers m-tiles {mg*2, mg*2+1}
    // (32 rows) and n-tiles {ng*4 .. ng*4+3} (64 cols). C = 8 WMMA tiles.
    const int mg = wave >> 2;
    const int ng = wave & 3;
    v8f acc[2][4];
#pragma unroll
    for (int i = 0; i < 2; ++i)
#pragma unroll
        for (int j = 0; j < 4; ++j) acc[i][j] = (v8f)0.0f;

    // Loop-invariant per-lane bases (computed once):
    //  - B: lane's 32B slot + this wave's n-group (ng*4 tiles * 1KB each).
    //  - A: lane's row-in-tile (M = lane&15) and K-half ((lane>>4)*16 f16).
    const char* bLane = (const char*)wB + (uint32_t)(lane * 32 + ng * 4 * 1024);
    const _Float16* aLane = sAgg + (uint32_t)((lane & 15) * D_IN + (lane >> 4) * 16);

    __syncthreads();

    for (int c0 = 0; c0 < R1; c0 += CR) {
        const int cR = (R1 - c0 < CR) ? (R1 - c0) : CR;

        // ---- zero the aggregation chunk ----
        {
            uint4 z = {0u, 0u, 0u, 0u};
            uint4* p = (uint4*)sAgg;
            const int n16 = CR * NODES_PER_BLOCK * D_IN * 2 / 16;
            for (int i = tid; i < n16; i += 256) p[i] = z;
        }
        __syncthreads();

        // ---- gather + accumulate: wave owns nodes [wave*8, wave*8+8) ----
        // Each edge's embedding row is fetched exactly once across all chunks
        // (only when its relation falls inside the current chunk).
        const int eBase = wave * (8 * S_EDGES);
        for (int e = 0; e < 8 * S_EDGES; ++e) {
            const int edge = eBase + e;
            const int r = sRel[edge] - c0;            // wave-uniform
            if ((unsigned)r < (unsigned)cR) {
                const int node = edge >> 5;           // node within block
                const int nbr  = sNbr[edge];
                const float* erow = emb + (size_t)nbr * D_IN + lane * 8;
                v4f f0 = *(const v4f*)erow;
                v4f f1 = *(const v4f*)(erow + 4);
                v8h h;
#pragma unroll
                for (int j = 0; j < 4; ++j) { h[j] = (_Float16)f0[j]; h[4 + j] = (_Float16)f1[j]; }
                v8h* dst = (v8h*)(sAgg + ((size_t)(r * NODES_PER_BLOCK + node) * D_IN) + lane * 8);
                *dst = *dst + h;                      // race-free: wave owns node
            }
        }
        __syncthreads();

        // ---- WMMA GEMM for this chunk: acc += A(agg_r) x B(W_r^T) ----
        // One 64-bit add per relation for B; all 64 loads per relation use
        // pure immediate offsets (max ~118KB, fits global IOFFSET[23:0]).
        // A offsets (mt*8KB + k*64B + {0,16}B <= ~25KB) fit DS's 16-bit imm.
        for (int r = 0; r < cR; ++r) {
            const int rg = c0 + r;
            const char* bR = bLane + (size_t)rg * B_REL_BYTES;
            const _Float16* aR = aLane + (uint32_t)r * (NODES_PER_BLOCK * D_IN)
                                       + (uint32_t)(mg * 2 * 16) * D_IN;
#pragma unroll
            for (int k = 0; k < KTILES; ++k) {
                v16h B[4];
#pragma unroll
                for (int j = 0; j < 4; ++j)
                    B[j] = *(const v16h*)(bR + (uint32_t)((k * NTILES + j) * 1024));
#pragma unroll
                for (int m = 0; m < 2; ++m) {
                    // A k order used here (same as packed B): v0-3 = K {half*16..+7},
                    //                            v4-7 = K {half*16+8..+15}
                    const _Float16* ab = aR + (uint32_t)(m * 16 * D_IN + k * 32);
                    union { v16h v; struct { v8h lo, hi; } s; } A;
                    A.s.lo = *(const v8h*)ab;
                    A.s.hi = *(const v8h*)(ab + 8);
#pragma unroll
                    for (int j = 0; j < 4; ++j)
                        acc[m][j] = __builtin_amdgcn_wmma_f32_16x16x32_f16(
                            false, A.v, false, B[j], (short)0, acc[m][j],
                            false, false);
                }
            }
        }
        __syncthreads();
    }

    // ---- epilogue: scale (mean/S, /R1, *1000), relu, store f32 ----
    // C layout: lanes 0-15 -> M = v, N = lane; lanes 16-31 -> M = v+8.
    // Single per-wave base pointer; all 64 stores use immediate offsets.
    const float scale = 1000.0f / (33.0f * 32.0f);
    float* outBase = out + (size_t)(blockNode0 + mg * 32 + ((lane >> 4) << 3)) * D_OUT
                         + (uint32_t)(ng * 64 + (lane & 15));
#pragma unroll
    for (int m = 0; m < 2; ++m) {
#pragma unroll
        for (int j = 0; j < 4; ++j) {
#pragma unroll
            for (int v = 0; v < 8; ++v) {
                float x = acc[m][j][v] * scale;
                x = x > 0.0f ? x : 0.0f;
                outBase[(uint32_t)((m * 16 + v) * D_OUT + j * 16)] = x;
            }
        }
    }
}

// =====================================================================
extern "C" void kernel_launch(void* const* d_in, const int* in_sizes, int n_in,
                              void* d_out, int out_size, void* d_ws, size_t ws_size,
                              hipStream_t stream) {
    const float* emb = (const float*)d_in[0];   // [100000, 256]
    const float* W   = (const float*)d_in[1];   // [33, 256, 256]
    const int*   nbr = (const int*)d_in[2];     // [8192, 32]
    const int*   rel = (const int*)d_in[3];     // [8192, 32]
    float*       out = (float*)d_out;           // [8192, 256]
    _Float16*    wB  = (_Float16*)d_ws;         // 33 * B_REL_BYTES = 4.33 MB

    const int prepThreads = R1 * KTILES * NTILES * 32;   // 135168
    prepack_weights_kernel<<<(prepThreads + 255) / 256, 256, 0, stream>>>(W, wB);

    rgcn_kernel<<<8192 / NODES_PER_BLOCK, 256, LDS_BYTES, stream>>>(
        emb, wB, nbr, rel, out);
}
